// StatelessConvLIF_88270167867787
// MI455X (gfx1250) — matrix operand order repaired and checked by reading, not compiled, not run on hardware
//
#include <hip/hip_runtime.h>
#include <hip/hip_bf16.h>

// ---------------------------------------------------------------------------
// Fused 3x3 conv (implicit GEMM via v_wmma_f32_16x16x32_f16) + LIF scan.
// x:  [T=8, B=16, Cin=64, H=64, W=64]  f32
// W:  [Cout=128, Cin=64, 3, 3]         f32
// out:[T, B, Cout, H, W]               f32 spikes (0.0 / 1.0)
//
// One block (256 threads = 8 wave32 waves) per (b, h) output row.
// Wave wv owns cout tile [wv*16, wv*16+16); the block's 64 pixels form
// 4 M-tiles of 16. The LIF scan over t is fused: membrane v[4] lives in
// VGPRs, the pre-activation z never touches memory.
// Per t: one barrier, WMMA from double-buffered LDS, stage t+1 (coalesced
// 264B runs, f32->f16 in regs), prefetch t+2, LIF + float4 spike stores.
// ---------------------------------------------------------------------------

typedef __attribute__((ext_vector_type(16))) _Float16 v16h;
typedef __attribute__((ext_vector_type(8)))  _Float16 v8h;
typedef __attribute__((ext_vector_type(8)))  float    v8f;

#define T_STEPS 8
#define BATCH   16
#define CIN     64
#define COUT    128
#define HH      64
#define WW      64
#define HW      (HH * WW)       // 4096
#define KTAPS   9
#define CINP    80              // padded ch extent: 160B (16B-aligned, bank-skewed)
#define XCOLS   66              // 64 cols + 1 halo each side

// Dynamic LDS (f16 units):
//   sW : [9 taps][128 cout][CINP]          = 92160 halves (184320 B)
//   sX : 2 x [3 rows][66 cols][CINP]       = 2*15840 halves (63360 B)
#define SW_HALVES   (KTAPS * COUT * CINP)
#define SXBUF_HALVES (3 * XCOLS * CINP)
#define SMEM_BYTES  ((SW_HALVES + 2 * SXBUF_HALVES) * 2)

__global__ __launch_bounds__(256)
void conv3x3_lif_wmma_kernel(const float* __restrict__ x,
                             const float* __restrict__ wgt,
                             float* __restrict__ out) {
    extern __shared__ _Float16 smem[];
    _Float16* sW = smem;                   // [tap][cout][CINP]
    _Float16* sX = smem + SW_HALVES;       // two buffers of [row][col][CINP]

    const unsigned tid  = threadIdx.x;
    const unsigned lane = tid & 31u;
    const unsigned wv   = tid >> 5;        // wave id -> cout tile
    const unsigned m    = lane & 15u;
    const unsigned hi   = lane >> 4;

    const unsigned b = blockIdx.x >> 6;    // batch 0..15
    const unsigned h = blockIdx.x & 63u;   // output row 0..63

    // ---- Stage weights once: f32 [cout][cin][3][3] -> f16 LDS [tap][cout][cin]
    for (unsigned idx = tid; idx < (unsigned)(KTAPS * COUT * CIN); idx += 256u) {
        const unsigned ci  = idx & 63u;
        const unsigned co  = (idx >> 6) & 127u;
        const unsigned tap = idx >> 13;
        sW[(tap * COUT + co) * CINP + ci] =
            (_Float16)wgt[co * (CIN * KTAPS) + ci * KTAPS + tap];
    }

    // ---- Stage x halo slice for timestep tt into buffer buf, prefetch tt+1.
    auto stage = [&](_Float16* buf, unsigned tt) {
        const float* xb = x + ((size_t)(tt * BATCH + b)) * (CIN * HW);
        const bool doPf = (tt + 1u) < T_STEPS;
        for (unsigned idx = tid; idx < (unsigned)(3 * XCOLS * CIN); idx += 256u) {
            const unsigned col = idx % (unsigned)XCOLS;      // 0..65, gw = col-1
            const unsigned rem = idx / (unsigned)XCOLS;
            const unsigned ci  = rem & 63u;
            const unsigned row = rem >> 6;                   // 0..2, gh = h+row-1
            const int gh = (int)h + (int)row - 1;
            const int gw = (int)col - 1;
            float val = 0.0f;
            if ((unsigned)gh < (unsigned)HH && (unsigned)gw < (unsigned)WW) {
                const size_t off = (size_t)ci * HW + (unsigned)gh * WW + (unsigned)gw;
                val = xb[off];
                if (doPf)  // pull next timestep toward L2 (global_prefetch_b8)
                    __builtin_prefetch(xb + (size_t)BATCH * CIN * HW + off, 0, 1);
            }
            buf[(row * (unsigned)XCOLS + col) * CINP + ci] = (_Float16)val;
        }
    };

    stage(sX, 0);   // prologue: buffer 0 holds t = 0

    // Membrane potential: 4 w-tiles x 8 accumulator elements per lane.
    v8f v0 = {}, v1 = {}, v2 = {}, v3 = {};
    v8f* vtile[4] = { &v0, &v1, &v2, &v3 };
    const unsigned coutIdx = wv * 16u + m;

    for (unsigned t = 0; t < T_STEPS; ++t) {
        __syncthreads();   // current buffer staged; prior reads drained
        const _Float16* cur = sX + (t & 1u) * SXBUF_HALVES;

        v8f a0 = {}, a1 = {}, a2 = {}, a3 = {};
        v8f* acc[4] = { &a0, &a1, &a2, &a3 };

        #pragma unroll
        for (unsigned tap = 0; tap < KTAPS; ++tap) {
            const unsigned r = tap / 3u;
            const unsigned s = tap - 3u * r;
            const _Float16* bBase = sW + ((tap * COUT + coutIdx) * CINP);
            #pragma unroll
            for (unsigned kc = 0; kc < 2u; ++kc) {
                // B fragment (32x16 f16): lanes 0-15 K {0..7,16..23}, lanes 16-31 K {8..15,24..31}
                const unsigned b0 = kc * 32u + hi * 8u;
                v16h bm;
                reinterpret_cast<v8h*>(&bm)[0] =
                    *reinterpret_cast<const v8h*>(bBase + b0);
                reinterpret_cast<v8h*>(&bm)[1] =
                    *reinterpret_cast<const v8h*>(bBase + b0 + 16u);

                // A fragment (16x32 f16): lanes 0-15 K {0..7,16..23},
                // lanes 16-31 K {8..15,24..31}; reuse B across 4 M-tiles.
                const unsigned a0off = kc * 32u + hi * 8u;
                #pragma unroll
                for (unsigned wt = 0; wt < 4u; ++wt) {
                    const _Float16* aBase =
                        cur + ((r * (unsigned)XCOLS + wt * 16u + m + s) * CINP);
                    v16h a;
                    reinterpret_cast<v8h*>(&a)[0] =
                        *reinterpret_cast<const v8h*>(aBase + a0off);
                    reinterpret_cast<v8h*>(&a)[1] =
                        *reinterpret_cast<const v8h*>(aBase + a0off + 16u);
                    *acc[wt] = __builtin_amdgcn_wmma_f32_16x16x32_f16(
                        false, a, false, bm, (short)0, *acc[wt], false, false);
                }
            }
        }

        // Stage next timestep into the other buffer (no barrier needed here:
        // it writes the buffer nobody reads until after the next barrier).
        if (t + 1u < T_STEPS)
            stage(sX + ((t + 1u) & 1u) * SXBUF_HALVES, t + 1u);

        // ---- Fused LIF + spike store per w-tile.
        #pragma unroll
        for (unsigned wt = 0; wt < 4u; ++wt) {
            float spk[8];
            v8f& vv = *vtile[wt];
            const v8f& zz = *acc[wt];
            #pragma unroll
            for (int e = 0; e < 8; ++e) {
                float vn = 0.5f * (vv[e] + zz[e]);          // v += (z - v)/tau
                float sp = (vn >= 1.0f) ? 1.0f : 0.0f;      // Heaviside(v - 1)
                spk[e] = sp;
                vv[e] = (sp != 0.0f) ? 0.0f : vn;           // hard reset to 0
            }
            // C/D layout: VGPR j, lane -> (M = j + 8*hi, N = lane&15):
            // consecutive j = consecutive w -> two aligned float4 stores.
            float* po = out
                + ((size_t)(t * BATCH + b) * COUT + coutIdx) * HW
                + h * WW + wt * 16u + hi * 8u;
            *reinterpret_cast<float4*>(po)     = *reinterpret_cast<const float4*>(&spk[0]);
            *reinterpret_cast<float4*>(po + 4) = *reinterpret_cast<const float4*>(&spk[4]);
        }
    }
}

extern "C" void kernel_launch(void* const* d_in, const int* in_sizes, int n_in,
                              void* d_out, int out_size, void* d_ws, size_t ws_size,
                              hipStream_t stream) {
    (void)in_sizes; (void)n_in; (void)out_size; (void)d_ws; (void)ws_size;
    const float* x   = (const float*)d_in[0];
    const float* wgt = (const float*)d_in[1];
    float* out       = (float*)d_out;

    dim3 grid(BATCH * HH);   // 1024 blocks: one (batch, row) each
    dim3 block(256);         // 8 wave32 waves
    conv3x3_lif_wmma_kernel<<<grid, block, SMEM_BYTES, stream>>>(x, wgt, out);
}
